// AttentionWithSharedWeights_90615220011811
// MI455X (gfx1250) — matrix operand order repaired and checked by reading, not compiled, not run on hardware
//
#include <hip/hip_runtime.h>
#include <hip/hip_bf16.h>

// Problem constants (match reference)
#define B_    4
#define S_    2048
#define DIM_  2048
#define NH_   16
#define NKV_  4
#define HD_   128
#define RANK_ 16
#define MTOT_ (B_*S_)   // 8192 rows

typedef __attribute__((ext_vector_type(16))) __bf16 bf16x16;
typedef __attribute__((ext_vector_type(8)))  __bf16 bf16x8;
typedef __attribute__((ext_vector_type(8)))  float  f32x8;

__device__ __forceinline__ bf16x16 cat8(bf16x8 lo, bf16x8 hi) {
  return __builtin_shufflevector(lo, hi, 0,1,2,3,4,5,6,7,8,9,10,11,12,13,14,15);
}

__device__ __forceinline__ f32x8 wmma_bf16(bf16x16 a, bf16x16 b, f32x8 c) {
  // D = A(16x32) * B(32x16) + C(16x16), f32 accumulate
  return __builtin_amdgcn_wmma_f32_16x16x32_bf16(false, a, false, b, (short)0, c, false, false);
}

// ---------------------------------------------------------------------------
// CDNA5 async global->LDS copy (ASYNCcnt-tracked, bypasses VGPRs).
// VDST VGPR carries the 32-bit LDS byte offset, VADDR the 64-bit global addr.
// ---------------------------------------------------------------------------
__device__ __forceinline__ void async_ld16(void* lds, const void* g) {
  unsigned loff = (unsigned)(unsigned long long)
      (__attribute__((address_space(3))) char*)lds;
  asm volatile("global_load_async_to_lds_b128 %0, %1, off"
               :: "v"(loff), "v"(g)
               : "memory");
}
__device__ __forceinline__ void wait_async_le4() {
  asm volatile("s_wait_asynccnt 0x4" ::: "memory");
}
__device__ __forceinline__ void wait_async_0() {
  asm volatile("s_wait_asynccnt 0x0" ::: "memory");
}

// ---------------------------------------------------------------------------
// f32 -> bf16 conversion
// ---------------------------------------------------------------------------
__global__ void k_cvt_bf16(const float* __restrict__ src, __bf16* __restrict__ dst, int n) {
  int i = blockIdx.x * blockDim.x + threadIdx.x;
  if (i < n) dst[i] = (__bf16)src[i];
}

// ---------------------------------------------------------------------------
// LoRA down-projection: T[m][r] = sum_k X[m][k] * A[r][k]   (A is [16, K] f32)
// ---------------------------------------------------------------------------
__global__ void k_lora_t(const __bf16* __restrict__ X, const float* __restrict__ A,
                         float* __restrict__ T, int K) {
  int idx = blockIdx.x * blockDim.x + threadIdx.x;   // MTOT_*RANK_ threads
  int r = idx & (RANK_ - 1);
  int m = idx >> 4;
  const __bf16* xr = X + (size_t)m * K;
  const float*  ar = A + (size_t)r * K;
  float acc = 0.f;
  for (int k = 0; k < K; k += 4) {
    acc += (float)xr[k+0] * ar[k+0];
    acc += (float)xr[k+1] * ar[k+1];
    acc += (float)xr[k+2] * ar[k+2];
    acc += (float)xr[k+3] * ar[k+3];
  }
  T[idx] = acc;
}

// ---------------------------------------------------------------------------
// WMMA GEMM with fused LoRA epilogue, async-LDS staged, double buffered.
//   C[m][n] = sum_k Xb[m][k]*Wb[n][k]  +  sum_r T[m][r]*Bm[n][r]
// Block = 128 threads (4 waves) -> 64x64 output tile; each wave owns a 32x32
// quadrant (2x2 WMMA accumulators). Per K-step (32), A/B 64x32 tiles are
// streamed into LDS via global_load_async_to_lds_b128 and re-read as WMMA
// fragments through ds_load.
// MODE selects the scatter layout:
//   0: Q  -> qh[b][h][s][d]   (bf16)
//   1: K  -> kh[b][kvh][s][d] (bf16)
//   2: V  -> vT[b][kvh][d][s] (bf16, transposed for attention B-fragments)
//   3: O  -> out[m][n]        (f32, final result)
// ---------------------------------------------------------------------------
template <int MODE>
__global__ __launch_bounds__(128)
void k_gemm_lora(const __bf16* __restrict__ Xb, const __bf16* __restrict__ Wb,
                 const float* __restrict__ T, const float* __restrict__ Bm,
                 void* __restrict__ outp, int K) {
  __shared__ __bf16 As[2][64][32];
  __shared__ __bf16 Bs[2][64][32];

  const int tid  = threadIdx.x;
  const int lane = tid & 31;
  const int wid  = tid >> 5;          // 0..3
  const int mi2  = wid >> 1;          // quadrant row (0/1)
  const int ni2  = wid & 1;           // quadrant col (0/1)
  const int l    = lane & 15;
  const int hl   = lane >> 4;
  const int aoff = hl * 8;            // A-frag K offset pattern
  const int boff = hl * 16;           // unused; B frags use aoff too
  const int m0 = blockIdx.y * 64;
  const int n0 = blockIdx.x * 64;

  // per-thread chunk assignment for async staging: 2x16B for A, 2x16B for B
  const int c0i = tid * 2, c1i = tid * 2 + 1;
  const int ar0 = c0i >> 2, ac0 = (c0i & 3) * 8;
  const int ar1 = c1i >> 2, ac1 = (c1i & 3) * 8;

  const int niter = K >> 5;

  // prologue: stage first K-tile into buffer 0
  {
    async_ld16(&As[0][ar0][ac0], Xb + (size_t)(m0 + ar0) * K + ac0);
    async_ld16(&As[0][ar1][ac1], Xb + (size_t)(m0 + ar1) * K + ac1);
    async_ld16(&Bs[0][ar0][ac0], Wb + (size_t)(n0 + ar0) * K + ac0);
    async_ld16(&Bs[0][ar1][ac1], Wb + (size_t)(n0 + ar1) * K + ac1);
  }

  f32x8 acc[2][2] = {};

  for (int i = 0; i < niter; ++i) {
    const int cur = i & 1;
    const int kn  = (i + 1) << 5;
    const bool has_next = (i + 1) < niter;
    if (has_next) {   // prefetch next K-tile into the other buffer
      async_ld16(&As[cur ^ 1][ar0][ac0], Xb + (size_t)(m0 + ar0) * K + kn + ac0);
      async_ld16(&As[cur ^ 1][ar1][ac1], Xb + (size_t)(m0 + ar1) * K + kn + ac1);
      async_ld16(&Bs[cur ^ 1][ar0][ac0], Wb + (size_t)(n0 + ar0) * K + kn + ac0);
      async_ld16(&Bs[cur ^ 1][ar1][ac1], Wb + (size_t)(n0 + ar1) * K + kn + ac1);
      wait_async_le4();   // async loads retire in order: current tile resident
    } else {
      wait_async_0();
    }
    __syncthreads();

    // WMMA fragments out of LDS
    const __bf16* a0r = &As[cur][mi2 * 32 + l][0];
    const __bf16* a1r = &As[cur][mi2 * 32 + 16 + l][0];
    const __bf16* b0r = &Bs[cur][ni2 * 32 + l][0];
    const __bf16* b1r = &Bs[cur][ni2 * 32 + 16 + l][0];
    bf16x16 a0 = cat8(*(const bf16x8*)(a0r + aoff), *(const bf16x8*)(a0r + aoff + 16));
    bf16x16 a1 = cat8(*(const bf16x8*)(a1r + aoff), *(const bf16x8*)(a1r + aoff + 16));
    bf16x16 b0 = cat8(*(const bf16x8*)(b0r + aoff), *(const bf16x8*)(b0r + aoff + 16));
    bf16x16 b1 = cat8(*(const bf16x8*)(b1r + aoff), *(const bf16x8*)(b1r + aoff + 16));
    acc[0][0] = wmma_bf16(a0, b0, acc[0][0]);
    acc[0][1] = wmma_bf16(a0, b1, acc[0][1]);
    acc[1][0] = wmma_bf16(a1, b0, acc[1][0]);
    acc[1][1] = wmma_bf16(a1, b1, acc[1][1]);

    __syncthreads();   // reads done before this buffer is overwritten again
  }

  // ---- fused LoRA epilogue + layout scatter (per-wave 32x32 quadrant) ----
  const int m0w = m0 + mi2 * 32;
  const int n0w = n0 + ni2 * 32;

  f32x8 bnA[2], bnB[2];
  for (int ni = 0; ni < 2; ++ni) {
    const f32x8* bp = (const f32x8*)(Bm + (size_t)(n0w + 16 * ni + l) * RANK_);
    bnA[ni] = bp[0];
    bnB[ni] = bp[1];
  }

  for (int mi = 0; mi < 2; ++mi) {
    for (int v = 0; v < 8; ++v) {
      const int m = m0w + 16 * mi + 8 * hl + v;          // global row (C layout)
      const f32x8* tp = (const f32x8*)(T + (size_t)m * RANK_);
      f32x8 ta = tp[0], tb = tp[1];
      for (int ni = 0; ni < 2; ++ni) {
        const int n = n0w + 16 * ni + l;                 // global col (C layout)
        f32x8 prod = ta * bnA[ni] + tb * bnB[ni];
        float lora = prod[0] + prod[1] + prod[2] + prod[3] +
                     prod[4] + prod[5] + prod[6] + prod[7];
        float val = acc[mi][ni][v] + lora;
        if (MODE == 0) {
          int b = m >> 11, s = m & (S_ - 1), h = n >> 7, d = n & (HD_ - 1);
          ((__bf16*)outp)[(((size_t)(b * NH_ + h)) * S_ + s) * HD_ + d] = (__bf16)val;
        } else if (MODE == 1) {
          int b = m >> 11, s = m & (S_ - 1), h = n >> 7, d = n & (HD_ - 1);
          ((__bf16*)outp)[(((size_t)(b * NKV_ + h)) * S_ + s) * HD_ + d] = (__bf16)val;
        } else if (MODE == 2) {
          int b = m >> 11, s = m & (S_ - 1), h = n >> 7, d = n & (HD_ - 1);
          ((__bf16*)outp)[(((size_t)(b * NKV_ + h)) * HD_ + d) * S_ + s] = (__bf16)val;
        } else {
          ((float*)outp)[(size_t)m * DIM_ + n] = val;
        }
      }
    }
  }
}

// ---------------------------------------------------------------------------
// RoPE, in place on head-major bf16 tensor [BH][S][HD]
// ---------------------------------------------------------------------------
__global__ void k_rope(__bf16* __restrict__ data, const float* __restrict__ cs,
                       const float* __restrict__ sn, int total) {
  int idx = blockIdx.x * blockDim.x + threadIdx.x;
  if (idx >= total) return;
  int d2 = idx & 63;                  // pair index within head dim
  int s  = (idx >> 6) & (S_ - 1);
  int bh = idx >> 17;                 // 64 * 2048 = 2^17 per (b,h)
  __bf16* p = data + ((size_t)bh * S_ + s) * HD_ + 2 * d2;
  float x0 = (float)p[0], x1 = (float)p[1];
  float c = cs[s * 64 + d2], si = sn[s * 64 + d2];
  p[0] = (__bf16)(x0 * c - x1 * si);
  p[1] = (__bf16)(x0 * si + x1 * c);
}

// ---------------------------------------------------------------------------
// Flash-style causal GQA attention. One wave per (b, h, 16-row q tile).
// Q fragments resident; loop over 32-key blocks; scores & PV via WMMA.
// P goes through LDS for the C-layout -> A-layout cross-lane transpose.
// ---------------------------------------------------------------------------
__global__ __launch_bounds__(32)
void k_attn(const __bf16* __restrict__ qh, const __bf16* __restrict__ kh,
            const __bf16* __restrict__ vT, __bf16* __restrict__ out) {
  const int lane = threadIdx.x;
  const int l  = lane & 15;
  const int hl = lane >> 4;
  const int aoff = hl * 8;
  const int boff = hl * 16;
  const int qt = blockIdx.x;      // q tile (16 rows)
  const int h  = blockIdx.y;
  const int b  = blockIdx.z;
  const int kvh = h >> 2;         // GQA: 16 heads -> 4 kv heads

  __shared__ __bf16 pl[16 * 32];

  // Resident Q fragments: A-layout, 4 chunks of K=32 over head_dim=128
  const __bf16* Qb = qh + (((size_t)(b * NH_ + h)) * S_ + qt * 16 + l) * HD_ + aoff;
  bf16x16 qf[4];
  for (int kc = 0; kc < 4; ++kc)
    qf[kc] = cat8(*(const bf16x8*)(Qb + kc * 32), *(const bf16x8*)(Qb + kc * 32 + 16));

  f32x8 o[8] = {};
  float rmax[8], rsum[8];
  for (int v = 0; v < 8; ++v) { rmax[v] = -1e30f; rsum[v] = 0.f; }

  const __bf16* Kb = kh + ((size_t)(b * NKV_ + kvh)) * S_ * HD_;
  const __bf16* Vb = vT + ((size_t)(b * NKV_ + kvh)) * HD_ * S_;
  const float scale = 0.08838834764831845f;  // 1/sqrt(128)
  const int nkb = (qt * 16 + 16 + 31) >> 5;  // causal: only key blocks <= q rows

  for (int kb = 0; kb < nkb; ++kb) {
    const int c0 = kb * 32;
    // S = Q * K^T for 2 column tiles of 16 keys
    f32x8 sc[2] = {};
    for (int j = 0; j < 2; ++j) {
      const __bf16* kr = Kb + (size_t)(c0 + 16 * j + l) * HD_ + aoff;
      for (int kc = 0; kc < 4; ++kc)
        sc[j] = wmma_bf16(qf[kc], cat8(*(const bf16x8*)(kr + kc * 32), *(const bf16x8*)(kr + kc * 32 + 16)), sc[j]);
    }
    // scale + causal mask + online softmax stats (rows striped per C layout)
    float nmax[8];
    for (int v = 0; v < 8; ++v) {
      int qrow = qt * 16 + 8 * hl + v;
      float s0 = sc[0][v] * scale; if (c0 + l      > qrow) s0 = -1e30f;
      float s1 = sc[1][v] * scale; if (c0 + 16 + l > qrow) s1 = -1e30f;
      sc[0][v] = s0; sc[1][v] = s1;
      float x = fmaxf(s0, s1);
      x = fmaxf(x, __shfl_xor(x, 1, 32));
      x = fmaxf(x, __shfl_xor(x, 2, 32));
      x = fmaxf(x, __shfl_xor(x, 4, 32));
      x = fmaxf(x, __shfl_xor(x, 8, 32));
      nmax[v] = fmaxf(rmax[v], x);
    }
    for (int v = 0; v < 8; ++v) {
      float corr = __expf(rmax[v] - nmax[v]);
      rmax[v] = nmax[v];
      rsum[v] *= corr;
      for (int n = 0; n < 8; ++n) o[n][v] *= corr;
      float p0 = __expf(sc[0][v] - nmax[v]);
      float p1 = __expf(sc[1][v] - nmax[v]);
      float ss = p0 + p1;
      ss += __shfl_xor(ss, 1, 32);
      ss += __shfl_xor(ss, 2, 32);
      ss += __shfl_xor(ss, 4, 32);
      ss += __shfl_xor(ss, 8, 32);
      rsum[v] += ss;
      pl[(8 * hl + v) * 32 + l]      = (__bf16)p0;
      pl[(8 * hl + v) * 32 + 16 + l] = (__bf16)p1;
    }
    __syncthreads();
    // Re-read P in A-fragment layout
    bf16x16 pf = cat8(*(const bf16x8*)(pl + l * 32 + aoff),
                      *(const bf16x8*)(pl + l * 32 + aoff + 16));
    __syncthreads();
    // O += P * V  (V stored transposed: vT[d][s] rows are contiguous over keys)
    for (int n = 0; n < 8; ++n) {
      const __bf16* vr = Vb + (size_t)(16 * n + l) * S_ + c0 + aoff;
      o[n] = wmma_bf16(pf, cat8(*(const bf16x8*)vr, *(const bf16x8*)(vr + 16)), o[n]);
    }
  }

  // finalize: divide by row sums, scatter to [b][s][h*128+d] bf16
  for (int v = 0; v < 8; ++v) {
    float inv = 1.0f / rsum[v];
    int row = qt * 16 + 8 * hl + v;
    __bf16* op = out + ((size_t)b * S_ + row) * (NH_ * HD_) + h * HD_ + l;
    for (int n = 0; n < 8; ++n)
      op[16 * n] = (__bf16)(o[n][v] * inv);
  }
}

// ---------------------------------------------------------------------------
// Host launcher
// ---------------------------------------------------------------------------
extern "C" void kernel_launch(void* const* d_in, const int* in_sizes, int n_in,
                              void* d_out, int out_size, void* d_ws, size_t ws_size,
                              hipStream_t stream) {
  const float* x  = (const float*)d_in[0];
  const float* fc = (const float*)d_in[1];
  const float* fs = (const float*)d_in[2];
  const float* wq = (const float*)d_in[3];
  const float* wk = (const float*)d_in[4];
  const float* wv = (const float*)d_in[5];
  const float* wo = (const float*)d_in[6];
  const float* aq = (const float*)d_in[7];
  const float* bq = (const float*)d_in[8];
  const float* ak = (const float*)d_in[9];
  const float* bk = (const float*)d_in[10];
  const float* av = (const float*)d_in[11];
  const float* bv = (const float*)d_in[12];
  const float* ao = (const float*)d_in[13];
  const float* bo = (const float*)d_in[14];
  float* outF = (float*)d_out;

  char* ws = (char*)d_ws;
  size_t off = 0;
  auto walloc = [&](size_t bytes) -> char* {
    char* p = ws + off;
    off += (bytes + 255) & ~(size_t)255;
    return p;
  };
  __bf16* xb   = (__bf16*)walloc((size_t)MTOT_ * DIM_ * 2);        // 32 MB
  __bf16* wqb  = (__bf16*)walloc((size_t)NH_ * HD_ * DIM_ * 2);    //  8 MB
  __bf16* wkb  = (__bf16*)walloc((size_t)NKV_ * HD_ * DIM_ * 2);   //  2 MB
  __bf16* wvb  = (__bf16*)walloc((size_t)NKV_ * HD_ * DIM_ * 2);   //  2 MB
  __bf16* wob  = (__bf16*)walloc((size_t)DIM_ * NH_ * HD_ * 2);    //  8 MB
  __bf16* qhB  = (__bf16*)walloc((size_t)B_ * NH_ * S_ * HD_ * 2); // 32 MB
  __bf16* khB  = (__bf16*)walloc((size_t)B_ * NKV_ * S_ * HD_ * 2);//  8 MB
  __bf16* vTB  = (__bf16*)walloc((size_t)B_ * NKV_ * HD_ * S_ * 2);//  8 MB
  __bf16* aoB  = (__bf16*)walloc((size_t)MTOT_ * NH_ * HD_ * 2);   // 32 MB
  float*  Tbuf = (float*)walloc((size_t)MTOT_ * RANK_ * 4);        // 0.5 MB

  auto cvt = [&](const float* s, __bf16* d, int n) {
    k_cvt_bf16<<<(n + 255) / 256, 256, 0, stream>>>(s, d, n);
  };
  cvt(x,  xb,  MTOT_ * DIM_);
  cvt(wq, wqb, NH_ * HD_ * DIM_);
  cvt(wk, wkb, NKV_ * HD_ * DIM_);
  cvt(wv, wvb, NKV_ * HD_ * DIM_);
  cvt(wo, wob, DIM_ * NH_ * HD_);

  const dim3 blk(128);
  const int tThreads = MTOT_ * RANK_;

  // Q projection (+LoRA), scatter to head-major
  k_lora_t<<<tThreads / 256, 256, 0, stream>>>(xb, aq, Tbuf, DIM_);
  k_gemm_lora<0><<<dim3((NH_ * HD_) / 64, MTOT_ / 64), blk, 0, stream>>>(
      xb, wqb, Tbuf, bq, qhB, DIM_);
  // K projection
  k_lora_t<<<tThreads / 256, 256, 0, stream>>>(xb, ak, Tbuf, DIM_);
  k_gemm_lora<1><<<dim3((NKV_ * HD_) / 64, MTOT_ / 64), blk, 0, stream>>>(
      xb, wkb, Tbuf, bk, khB, DIM_);
  // V projection (stored transposed)
  k_lora_t<<<tThreads / 256, 256, 0, stream>>>(xb, av, Tbuf, DIM_);
  k_gemm_lora<2><<<dim3((NKV_ * HD_) / 64, MTOT_ / 64), blk, 0, stream>>>(
      xb, wvb, Tbuf, bv, vTB, DIM_);

  // RoPE on q and k
  int nq = B_ * NH_ * S_ * 64;
  k_rope<<<(nq + 255) / 256, 256, 0, stream>>>(qhB, fc, fs, nq);
  int nk = B_ * NKV_ * S_ * 64;
  k_rope<<<(nk + 255) / 256, 256, 0, stream>>>(khB, fc, fs, nk);

  // Attention
  k_attn<<<dim3(S_ / 16, NH_, B_), dim3(32), 0, stream>>>(qhB, khB, vTB, aoB);

  // Output projection (+LoRA), f32 result
  k_lora_t<<<tThreads / 256, 256, 0, stream>>>(aoB, ao, Tbuf, NH_ * HD_);
  k_gemm_lora<3><<<dim3(DIM_ / 64, MTOT_ / 64), blk, 0, stream>>>(
      aoB, wob, Tbuf, bo, outF, NH_ * HD_);
}
